// InfoNCE_55619826483436
// MI455X (gfx1250) — matrix operand order repaired and checked by reading, not compiled, not run on hardware
//
#include <hip/hip_runtime.h>
#include <hip/hip_bf16.h>

// ---------------------------------------------------------------------------
// InfoNCE fused: normalize -> f16 WMMA GEMM (z1n@z2n^T | z1n@z3n^T) -> fused
// exp-sum (no max needed: logits bounded to [-20,20]) -> mean(log S - diag).
// Round 2: double-buffered B-tile prefetch, branch-free hot loop, hoisted
// pos/neg base select, diagonal tile recomputed outside the loop.
// ---------------------------------------------------------------------------

typedef __attribute__((ext_vector_type(16))) _Float16 v16h;
typedef __attribute__((ext_vector_type(8)))  _Float16 v8h;
typedef __attribute__((ext_vector_type(8)))  float    v8f;
typedef __attribute__((ext_vector_type(4)))  float    v4f;

#define BROWS     4096
#define EDIM      256
#define INV_TEMP  20.0f          // 1 / 0.05
#define NTILES    512            // 256 pos + 256 neg column tiles of 16
#define NSPLIT    8
#define TPW       (NTILES / NSPLIT)   // 64 N-tiles per wave
#define TILE_STRIDE (16 * EDIM)       // halfs between consecutive N-tiles

// ---------------------------------------------------------------------------
// Kernel 1: L2-normalize rows of z1/z2/z3 (fp32) -> f16 rows in workspace.
// One wave per row (32 lanes x 8 elems). Also zeroes the per-row S array.
// grid = (512, 3), block = 256 (8 waves -> 8 rows per block).
// ---------------------------------------------------------------------------
__global__ __launch_bounds__(256) void infonce_normalize(
    const float* __restrict__ z1, const float* __restrict__ z2,
    const float* __restrict__ z3,
    _Float16* __restrict__ An, _Float16* __restrict__ Bn,
    _Float16* __restrict__ Cn, float* __restrict__ S)
{
    const int lane = threadIdx.x & 31;
    const int wave = threadIdx.x >> 5;
    const int row  = blockIdx.x * 8 + wave;

    const float* src = (blockIdx.y == 0) ? z1 : ((blockIdx.y == 1) ? z2 : z3);
    _Float16*    dst = (blockIdx.y == 0) ? An : ((blockIdx.y == 1) ? Bn : Cn);

    const float* p = src + (size_t)row * EDIM + lane * 8;
    v4f a = *(const v4f*)p;
    v4f b = *(const v4f*)(p + 4);

    float ss = a.x*a.x + a.y*a.y + a.z*a.z + a.w*a.w
             + b.x*b.x + b.y*b.y + b.z*b.z + b.w*b.w;
    ss += __shfl_xor(ss, 1, 32);
    ss += __shfl_xor(ss, 2, 32);
    ss += __shfl_xor(ss, 4, 32);
    ss += __shfl_xor(ss, 8, 32);
    ss += __shfl_xor(ss, 16, 32);

    const float inv = 1.0f / fmaxf(sqrtf(ss), 1e-12f);

    v8h h;
    h[0] = (_Float16)(a.x * inv);  h[1] = (_Float16)(a.y * inv);
    h[2] = (_Float16)(a.z * inv);  h[3] = (_Float16)(a.w * inv);
    h[4] = (_Float16)(b.x * inv);  h[5] = (_Float16)(b.y * inv);
    h[6] = (_Float16)(b.z * inv);  h[7] = (_Float16)(b.w * inv);
    *(v8h*)(dst + (size_t)row * EDIM + lane * 8) = h;

    // zero the per-row exp-sum accumulator once (gridDim.y slice 0 only)
    if (blockIdx.y == 0) {
        int g = blockIdx.x * 256 + threadIdx.x;
        if (g < BROWS) S[g] = 0.0f;
    }
}

// ---- B-tile fragment load: B 32x16 layout, lane = column, 16 contig K -----
__device__ __forceinline__ void load_btile(v16h* __restrict__ bf,
                                           const _Float16* __restrict__ p)
{
#pragma unroll
    for (int kc = 0; kc < 8; ++kc) {
        union { v16h v; v8h h[2]; } u;
        u.h[0] = *(const v8h*)(p + kc * 32);
        u.h[1] = *(const v8h*)(p + kc * 32 + 8);
        bf[kc] = u.v;
    }
}

// ---- one 16x16 tile: K=256 WMMA chain --------------------------------------
__device__ __forceinline__ v8f tile_mma(const v16h* __restrict__ afrag,
                                        const v16h* __restrict__ bf)
{
    v8f acc = {};
#pragma unroll
    for (int kc = 0; kc < 8; ++kc)
        acc = __builtin_amdgcn_wmma_f32_16x16x32_f16(
            false, afrag[kc], false, bf[kc], (short)0, acc, false, false);
    return acc;
}

// ---------------------------------------------------------------------------
// Kernel 2: fused GEMM + exp-sum. One wave = one 16-row M-tile x 64 N-tiles.
// grid = 256 blocks (one M-tile each), block = 256 (8 waves = 8-way N split;
// waves 0-3 cover the pos half, waves 4-7 the neg half -> base ptr uniform).
// ---------------------------------------------------------------------------
__global__ __launch_bounds__(256) void infonce_gemm_softmax(
    const _Float16* __restrict__ An, const _Float16* __restrict__ Bn,
    const _Float16* __restrict__ Cn,
    float* __restrict__ S, float* __restrict__ D)
{
    const int lane  = threadIdx.x & 31;
    const int wave  = threadIdx.x >> 5;
    const int mtile = blockIdx.x;          // 0..255
    const int M0    = mtile * 16;
    const int nt0   = wave * TPW;

    const int hsel = lane >> 4;            // 0: lanes 0-15, 1: lanes 16-31

    // ---- preload A fragments for all 8 K-chunks (contiguous k order, same as load_btile)
    // lane<16 holds K {c*32+0..15}; lane>=16 holds {c*32+16..31}
    const _Float16* arow = An + (size_t)(M0 + (lane & 15)) * EDIM;
    v16h afrag[8];
#pragma unroll
    for (int kc = 0; kc < 8; ++kc) {
        const _Float16* p = arow + kc * 32 + hsel * 16;
        union { v16h v; v8h h[2]; } u;
        u.h[0] = *(const v8h*)(p);
        u.h[1] = *(const v8h*)(p + 8);
        afrag[kc] = u.v;
    }

    // whole wave range lies in one source matrix (TPW=64 divides 256)
    const _Float16* Bsrc  = (nt0 < 256) ? Bn : Cn;
    const int       lnt0  = nt0 & 255;
    const _Float16* bptr  = Bsrc
        + (size_t)(lnt0 * 16 + (lane & 15)) * EDIM + hsel * 16;

    float rowsum[8];
#pragma unroll
    for (int v = 0; v < 8; ++v) rowsum[v] = 0.0f;

    // ---- double-buffered main loop: prefetch tile t+1 while computing t ----
    v16h bf0[8], bf1[8];
    load_btile(bf0, bptr);

    for (int t = 0; t < TPW; t += 2) {
        load_btile(bf1, bptr + (size_t)(t + 1) * TILE_STRIDE);

        v8f acc = tile_mma(afrag, bf0);
#pragma unroll
        for (int v = 0; v < 8; ++v) rowsum[v] += __expf(acc[v] * INV_TEMP);

        const int tn = (t + 2 < TPW) ? (t + 2) : t;   // clamped tail prefetch
        load_btile(bf0, bptr + (size_t)tn * TILE_STRIDE);

        acc = tile_mma(afrag, bf1);
#pragma unroll
        for (int v = 0; v < 8; ++v) rowsum[v] += __expf(acc[v] * INV_TEMP);
    }

    // reduce row sums across each 16-lane half (xor 1,2,4,8 stay in-half)
#pragma unroll
    for (int v = 0; v < 8; ++v) {
        float s = rowsum[v];
        s += __shfl_xor(s, 1, 32);
        s += __shfl_xor(s, 2, 32);
        s += __shfl_xor(s, 4, 32);
        s += __shfl_xor(s, 8, 32);
        rowsum[v] = s;
    }

    // lanes 0 / 16 own rows M0+0..7 / M0+8..15
    if (lane == 0 || lane == 16) {
        const int rbase = M0 + (lane >> 4) * 8;
#pragma unroll
        for (int v = 0; v < 8; ++v)
            atomicAdd(&S[rbase + v], rowsum[v]);
    }

    // ---- diagonal: recompute the single diag tile of sim_pos outside the
    // hot loop (only the wave whose N-range covers it; neg waves never match)
    if (mtile >= nt0 && mtile < nt0 + TPW) {
        const _Float16* dp = Bn
            + (size_t)(mtile * 16 + (lane & 15)) * EDIM + hsel * 16;
        v16h bfd[8];
        load_btile(bfd, dp);
        v8f acc = tile_mma(afrag, bfd);

        // lane v (v<8) holds diag of row v in acc[v];
        // lane 24+v holds diag of row 8+v in acc[v]
        const int idx = (lane < 8) ? lane : (lane - 24);
        float dv = 0.0f;
#pragma unroll
        for (int v = 0; v < 8; ++v)
            if (v == idx) dv = acc[v];
        const float diag = dv * INV_TEMP;

        if (lane < 8)        D[M0 + lane]      = diag;
        else if (lane >= 24) D[M0 + lane - 16] = diag;
    }
}

// ---------------------------------------------------------------------------
// Kernel 3: loss = mean_r( log(S_r) - diag_r ). Single block of 256.
// ---------------------------------------------------------------------------
__global__ __launch_bounds__(256) void infonce_finalize(
    const float* __restrict__ S, const float* __restrict__ D,
    float* __restrict__ out)
{
    __shared__ float sh[8];
    const int lane = threadIdx.x & 31;
    const int wave = threadIdx.x >> 5;

    float t = 0.0f;
    for (int r = threadIdx.x; r < BROWS; r += 256)
        t += __logf(S[r]) - D[r];

    t += __shfl_xor(t, 1, 32);
    t += __shfl_xor(t, 2, 32);
    t += __shfl_xor(t, 4, 32);
    t += __shfl_xor(t, 8, 32);
    t += __shfl_xor(t, 16, 32);
    if (lane == 0) sh[wave] = t;
    __syncthreads();
    if (threadIdx.x == 0) {
        float tot = 0.0f;
#pragma unroll
        for (int w = 0; w < 8; ++w) tot += sh[w];
        out[0] = tot * (1.0f / (float)BROWS);
    }
}

// ---------------------------------------------------------------------------
extern "C" void kernel_launch(void* const* d_in, const int* in_sizes, int n_in,
                              void* d_out, int out_size, void* d_ws, size_t ws_size,
                              hipStream_t stream) {
    const float* z1 = (const float*)d_in[0];
    const float* z2 = (const float*)d_in[1];
    const float* z3 = (const float*)d_in[2];

    // workspace layout: S[4096] f32 | D[4096] f32 (@16KB) | An|Bn|Cn f16 (@32KB)
    float*    S  = (float*)d_ws;
    float*    D  = (float*)((char*)d_ws + 16384);
    _Float16* An = (_Float16*)((char*)d_ws + 32768);
    _Float16* Bn = An + (size_t)BROWS * EDIM;
    _Float16* Cn = Bn + (size_t)BROWS * EDIM;

    infonce_normalize<<<dim3(512, 3), 256, 0, stream>>>(z1, z2, z3, An, Bn, Cn, S);
    infonce_gemm_softmax<<<256, 256, 0, stream>>>(An, Bn, Cn, S, D);
    infonce_finalize<<<1, 256, 0, stream>>>(S, D, (float*)d_out);
}
